// TorchCrepe2_63359357550812
// MI455X (gfx1250) — matrix operand order repaired and checked by reading, not compiled, not run on hardware
//
#include <hip/hip_runtime.h>
#include <hip/hip_bf16.h>
#include <math.h>

typedef __bf16 bf16_t;
typedef __attribute__((ext_vector_type(8)))  float  v8f;
typedef __attribute__((ext_vector_type(16))) __bf16 v16bf;
typedef __attribute__((ext_vector_type(8)))  __bf16 v8bf;

#define N_FRAMES    201
#define WINDOW_     1024
#define HOP_        160
#define NSAMP_      32000
#define BN_EPS_     0.0010000000474974513f
#define PITCH_BINS_ 360
#define CENTS_OFF_  1997.379408437619

#if __has_builtin(__builtin_amdgcn_global_load_async_to_lds_b128) && \
    __has_builtin(__builtin_amdgcn_s_wait_asynccnt)
#define HAVE_ASYNC_LDS 1
// Builtin signature (from hipcc diagnostic): arg0 = global-AS pointer to
// int __attribute__((vector_size(16))), arg1 = LDS-AS pointer, then 2 imms.
typedef __attribute__((vector_size(16))) int v4i32;
typedef __attribute__((address_space(1))) v4i32* gas_v4i_p;
typedef __attribute__((address_space(3))) v4i32* las_v4i_p;
#else
#define HAVE_ASYNC_LDS 0
#endif

// ---------------------------------------------------------------------------
// Frame extraction + per-frame mean/std (ddof=1) normalization -> bf16 frames
// ---------------------------------------------------------------------------
__global__ __launch_bounds__(256) void frames_kernel(const float* __restrict__ audio,
                                                     bf16_t* __restrict__ frames)
{
    __shared__ float red[256];
    const int t = blockIdx.x, tid = threadIdx.x;
    float vals[4];
    float s = 0.f;
#pragma unroll
    for (int i = 0; i < 4; ++i) {
        int j = tid + i * 256;
        int p = t * HOP_ + j - (WINDOW_ / 2);
        float v = (p >= 0 && p < NSAMP_) ? audio[p] : 0.f;
        vals[i] = v; s += v;
    }
    red[tid] = s; __syncthreads();
    for (int off = 128; off > 0; off >>= 1) {
        if (tid < off) red[tid] += red[tid + off];
        __syncthreads();
    }
    const float mean = red[0] / (float)WINDOW_;
    __syncthreads();
    float ss = 0.f;
#pragma unroll
    for (int i = 0; i < 4; ++i) { float d = vals[i] - mean; ss += d * d; }
    red[tid] = ss; __syncthreads();
    for (int off = 128; off > 0; off >>= 1) {
        if (tid < off) red[tid] += red[tid + off];
        __syncthreads();
    }
    const float stdv = sqrtf(red[0] / (float)(WINDOW_ - 1));
    const float inv  = 1.f / fmaxf(1e-10f, stdv);
#pragma unroll
    for (int i = 0; i < 4; ++i) {
        int j = tid + i * 256;
        frames[t * WINDOW_ + j] = (bf16_t)((vals[i] - mean) * inv);
    }
}

// ---------------------------------------------------------------------------
// fp32 -> bf16 conversion (weights)
// ---------------------------------------------------------------------------
__global__ __launch_bounds__(256) void cvt_bf16_kernel(const float* __restrict__ x,
                                                       bf16_t* __restrict__ y, int n)
{
    int i = blockIdx.x * 256 + threadIdx.x;
    if (i < n) y[i] = (bf16_t)x[i];
}

// ---------------------------------------------------------------------------
// Implicit-im2col conv1d as WMMA GEMM + fused bias/ReLU/BN/maxpool2 epilogue.
//   M = F*Lout, N = Cout, Kdim = Cin*Kw.
//   A[row,k] = act[f, k>>KWS, l*STRIDE - padL + (k & (Kw-1))] (0 if OOB)
//   B[k,n]   = w[n*Kdim + k]
// Block: 256 thr = 8 waves; macro-tile 64(M) x 128(N); BK=64 (2 wmma K-steps).
// Wave: 16(M) x 64(N) = 4 accumulators -> 8 v_wmma per K-macro-step per wave.
// ---------------------------------------------------------------------------
template<int KWS, int LOS, int STRIDE>
__global__ __launch_bounds__(256) void conv_wmma_kernel(
    const bf16_t* __restrict__ act, const bf16_t* __restrict__ wB,
    const float* __restrict__ bias, const float* __restrict__ gamma,
    const float* __restrict__ beta, const float* __restrict__ mean,
    const float* __restrict__ var,  bf16_t* __restrict__ out,
    int F, int Cin, int Lin, int Cout, int padL)
{
    const int Kw   = 1 << KWS;
    const int Lout = 1 << LOS;
    const int BM = 64, BN = 128, BK = 64, LDK = 72;   // 144B row stride (16B aligned)
    __shared__ bf16_t lA[BM][LDK];
    __shared__ bf16_t lB[BN][LDK];

    const int M    = F << LOS;
    const int Kdim = Cin << KWS;
    const int m0   = blockIdx.x * BM;
    const int n0   = blockIdx.y * BN;
    const int tid  = threadIdx.x;
    const int lane = tid & 31;
    const int wv   = tid >> 5;
    const int wm   = (wv & 3) << 4;      // 0,16,32,48
    const int wn   = (wv >> 2) << 6;     // 0,64

    v8f acc[4];
#pragma unroll
    for (int i = 0; i < 4; ++i) acc[i] = (v8f){0.f,0.f,0.f,0.f,0.f,0.f,0.f,0.f};

    // A fill: 4 threads per row, 16 elems each (2 chunks of 8)
    const int arow  = tid >> 2;          // 0..63
    const int ak0   = (tid & 3) << 4;    // 0,16,32,48
    const int gm    = m0 + arow;
    const int fA    = gm >> LOS;
    const int baseL = (gm & (Lout - 1)) * STRIDE - padL;
    const bf16_t* actRow = act + (size_t)fA * Cin * Lin;

    // B fill: 2 threads per row, 32 elems each (4 chunks of 8)
    const int brow = tid >> 1;           // 0..127
    const int bk0  = (tid & 1) << 5;     // 0,32
    const int bn   = n0 + brow;
    const bf16_t* wRow = wB + (size_t)bn * Kdim;

    for (int k0 = 0; k0 < Kdim; k0 += BK) {
        // ---- A tile (implicit im2col gather; vector fast path) ----
#pragma unroll
        for (int c = 0; c < 2; ++c) {
            const int kg0 = k0 + ak0 + c * 8;
            const int ci  = kg0 >> KWS;          // chunk of 8 never crosses ci
            const int kk0 = kg0 & (Kw - 1);
            const int lp0 = baseL + kk0;
            if (gm < M && lp0 >= 0 && lp0 + 7 < Lin) {
                v8bf v;
                __builtin_memcpy(&v, actRow + ci * Lin + lp0, 16);  // 2B-aligned wide load
                *(v8bf*)&lA[arow][ak0 + c * 8] = v;                  // ds_store_b128
            } else {
#pragma unroll
                for (int j = 0; j < 8; ++j) {
                    int lp = lp0 + j;
                    bf16_t e = (bf16_t)0.0f;
                    if (gm < M && lp >= 0 && lp < Lin) e = actRow[ci * Lin + lp];
                    lA[arow][ak0 + c * 8 + j] = e;
                }
            }
        }
        // ---- B tile (weights; aligned, always in-bounds) ----
#if HAVE_ASYNC_LDS
#pragma unroll
        for (int c = 0; c < 4; ++c) {
            __builtin_amdgcn_global_load_async_to_lds_b128(
                (gas_v4i_p)(void*)(wRow + k0 + bk0 + c * 8),
                (las_v4i_p)(void*)&lB[brow][bk0 + c * 8],
                0, 0);
        }
#else
#pragma unroll
        for (int c = 0; c < 4; ++c) {
            v8bf v = *(const v8bf*)(wRow + k0 + bk0 + c * 8);
            *(v8bf*)&lB[brow][bk0 + c * 8] = v;
        }
#endif
        if (k0 + BK < Kdim)
            __builtin_prefetch(wRow + k0 + BK + bk0, 0, 1);   // global_prefetch_b8
#if HAVE_ASYNC_LDS
        __builtin_amdgcn_s_wait_asynccnt(0);
#endif
        __syncthreads();

        // ---- compute: 2 K-substeps x 4 N-tiles ----
#pragma unroll
        for (int s = 0; s < 2; ++s) {
            union Frag { v16bf v; v8bf h[2]; };
            // A frag (16x32): lane m<16 -> row m, K 0..15; lane m+16 -> K 16..31
            const int ar = wm + (lane & 15);
            const int ka = s * 32 + ((lane < 16) ? 0 : 16);
            Frag Af;
            Af.h[0] = *(const v8bf*)&lA[ar][ka];
            Af.h[1] = *(const v8bf*)&lA[ar][ka + 8];
            // B frag (32x16): lane c<16 -> col c, K 0..15; lane c+16 -> K 16..31
            const int kb = s * 32 + ((lane < 16) ? 0 : 16);
#pragma unroll
            for (int t2 = 0; t2 < 4; ++t2) {
                const int bc = wn + t2 * 16 + (lane & 15);
                Frag Bf;
                Bf.h[0] = *(const v8bf*)&lB[bc][kb];
                Bf.h[1] = *(const v8bf*)&lB[bc][kb + 8];
                acc[t2] = __builtin_amdgcn_wmma_f32_16x16x32_bf16(
                    false, Af.v, false, Bf.v, (short)0, acc[t2], false, false);
            }
        }
        __syncthreads();
    }

    // ---- epilogue: bias + ReLU + BN + in-register maxpool2 + bf16 store ----
    // C layout: lane<16: N=lane, VGPR r -> M=r; lane>=16: N=lane-16, M=8+r.
    const int ncol  = lane & 15;
    const int mbase = (lane >> 4) << 3;
    const int halfL = Lout >> 1;
#pragma unroll
    for (int t2 = 0; t2 < 4; ++t2) {
        const int co = n0 + wn + t2 * 16 + ncol;
        if (co < Cout) {
            float bia = bias[co];
            float s   = gamma[co] * rsqrtf(var[co] + BN_EPS_);
            float mu  = mean[co], bb = beta[co];
#pragma unroll
            for (int r = 0; r < 4; ++r) {
                float x0 = fmaxf(acc[t2][2 * r]     + bia, 0.f);
                float x1 = fmaxf(acc[t2][2 * r + 1] + bia, 0.f);
                x0 = (x0 - mu) * s + bb;
                x1 = (x1 - mu) * s + bb;
                float pooled = fmaxf(x0, x1);
                int g = m0 + wm + mbase + 2 * r;       // even global row
                if (g < M) {
                    int f  = g >> LOS;
                    int l2 = (g & (Lout - 1)) >> 1;
                    out[((size_t)f * Cout + co) * halfL + l2] = (bf16_t)pooled;
                }
            }
        }
    }
}

// ---------------------------------------------------------------------------
// Classifier: probs[t,bin] = sigmoid( flat(transpose(a5)) . W[bin] + b[bin] )
// a5 layout (F, 512, 4); flatten order k = l*512 + c.
// ---------------------------------------------------------------------------
__global__ __launch_bounds__(256) void classifier_kernel(
    const bf16_t* __restrict__ a5, const float* __restrict__ W,
    const float* __restrict__ b, float* __restrict__ probs)
{
    int idx = blockIdx.x * 256 + threadIdx.x;
    if (idx >= N_FRAMES * PITCH_BINS_) return;
    int t   = idx / PITCH_BINS_;
    int bin = idx - t * PITCH_BINS_;
    const bf16_t* ap = a5 + t * 2048;
    const float*  wp = W + bin * 2048;
    float sum = b[bin];
    for (int k = 0; k < 2048; ++k) {
        int l = k >> 9, c = k & 511;
        sum += (float)ap[c * 4 + l] * wp[k];
    }
    probs[idx] = 1.f / (1.f + expf(-sum));
}

// ---------------------------------------------------------------------------
// JAX threefry2x32-20 replication for _dither(cents, jax.random.key(123)).
// ---------------------------------------------------------------------------
__device__ __forceinline__ void tf_mix(unsigned& x0, unsigned& x1, int r)
{
    x0 += x1; x1 = (x1 << r) | (x1 >> (32 - r)); x1 ^= x0;
}

__global__ __launch_bounds__(256) void dither_weights_kernel(float* __restrict__ wts)
{
    int i = blockIdx.x * 256 + threadIdx.x;
    if (i >= PITCH_BINS_) return;
    const bool second = (i >= 180);
    unsigned x0 = second ? (unsigned)(i - 180) : (unsigned)i;
    unsigned x1 = x0 + 180u;
    const unsigned k0 = 0u, k1 = 123u, k2 = k0 ^ k1 ^ 0x1BD11BDAu;
    x0 += k0; x1 += k1;
    tf_mix(x0,x1,13); tf_mix(x0,x1,15); tf_mix(x0,x1,26); tf_mix(x0,x1,6);
    x0 += k1; x1 += k2 + 1u;
    tf_mix(x0,x1,17); tf_mix(x0,x1,29); tf_mix(x0,x1,16); tf_mix(x0,x1,24);
    x0 += k2; x1 += k0 + 2u;
    tf_mix(x0,x1,13); tf_mix(x0,x1,15); tf_mix(x0,x1,26); tf_mix(x0,x1,6);
    x0 += k0; x1 += k1 + 3u;
    tf_mix(x0,x1,17); tf_mix(x0,x1,29); tf_mix(x0,x1,16); tf_mix(x0,x1,24);
    x0 += k1; x1 += k2 + 4u;
    tf_mix(x0,x1,13); tf_mix(x0,x1,15); tf_mix(x0,x1,26); tf_mix(x0,x1,6);
    x0 += k2; x1 += k0 + 5u;
    unsigned bits = second ? x1 : x0;
    float u = __uint_as_float((bits >> 9) | 0x3f800000u) - 1.0f;
    const float c = 0.5f, loc = -20.f, scale = 40.f;
    float f = (u < c) ? (c - u) / (scale * 0.5f) : (u - c) / (scale * 0.5f);
    float noise = 2.f * scale * sqrtf(1.f - fabsf(f)) + loc;
    if (u >= c) noise = 2.f * (scale - noise);
    wts[i] = 20.f * (float)i + (float)CENTS_OFF_ + noise;
}

// ---------------------------------------------------------------------------
// Decode: band-limit, argmax, 9-bin window, sigmoid-weighted cents -> pitch/period.
// ---------------------------------------------------------------------------
__global__ __launch_bounds__(64) void postproc_kernel(
    const float* __restrict__ probs, const float* __restrict__ wts,
    const int* __restrict__ pf0min, const int* __restrict__ pf0max,
    float* __restrict__ out)
{
    __shared__ float sval[64];
    __shared__ int   sidx[64];
    const int t = blockIdx.x, tid = threadIdx.x;
    const double fmin = (double)(*pf0min), fmax = (double)(*pf0max);
    const int minidx = (int)floor((1200.0 * log2(fmin / 10.0) - CENTS_OFF_) / 20.0);
    const int maxidx = (int)ceil ((1200.0 * log2(fmax / 10.0) - CENTS_OFF_) / 20.0);

    float best = -INFINITY; int bidx = 1 << 30;
    for (int j = tid; j < PITCH_BINS_; j += 64) {
        float v = (j >= minidx && j < maxidx) ? probs[t * PITCH_BINS_ + j] : -INFINITY;
        if (v > best) { best = v; bidx = j; }
    }
    sval[tid] = best; sidx[tid] = bidx;
    __syncthreads();
    if (tid == 0) {
        float bv = -INFINITY; int bi = 1 << 30;
        for (int i = 0; i < 64; ++i)
            if (sval[i] > bv || (sval[i] == bv && sidx[i] < bi)) { bv = sval[i]; bi = sidx[i]; }
        float num = 0.f, den = 0.f;
        int lo = bi - 4 < 0 ? 0 : bi - 4;
        int hi = bi + 5 > PITCH_BINS_ ? PITCH_BINS_ : bi + 5;
        for (int j = lo; j < hi; ++j) {
            if (j >= minidx && j < maxidx) {
                float lg = probs[t * PITCH_BINS_ + j];
                float p  = 1.f / (1.f + expf(-lg));
                num += wts[j] * p; den += p;
            }
        }
        float cents = num / den;
        out[t]            = 10.f * exp2f(cents / 1200.f);   // pitch
        out[N_FRAMES + t] = bv;                             // period = masked@argmax
    }
}

// ---------------------------------------------------------------------------
// Host launcher. Input flatten order (jax tree, sorted dict keys):
//   0: audio; 1..24: bns = 6 x (g,be,m,v); 25: cls_b; 26: cls_w;
//   27..38: convs = 6 x (w,b); 39: f0_min; 40: f0_max
// ---------------------------------------------------------------------------
extern "C" void kernel_launch(void* const* d_in, const int* in_sizes, int n_in,
                              void* d_out, int out_size, void* d_ws, size_t ws_size,
                              hipStream_t stream)
{
    (void)in_sizes; (void)n_in; (void)out_size; (void)ws_size;

    const float* audio = (const float*)d_in[0];
    const float *bn_g[6], *bn_be[6], *bn_m[6], *bn_v[6];
    for (int i = 0; i < 6; ++i) {
        bn_g[i]  = (const float*)d_in[1 + i * 4 + 0];
        bn_be[i] = (const float*)d_in[1 + i * 4 + 1];
        bn_m[i]  = (const float*)d_in[1 + i * 4 + 2];
        bn_v[i]  = (const float*)d_in[1 + i * 4 + 3];
    }
    const float* cls_b = (const float*)d_in[25];
    const float* cls_w = (const float*)d_in[26];
    const float *cw[6], *cb[6];
    for (int i = 0; i < 6; ++i) {
        cw[i] = (const float*)d_in[27 + 2 * i];
        cb[i] = (const float*)d_in[28 + 2 * i];
    }
    const int* f0min = (const int*)d_in[39];
    const int* f0max = (const int*)d_in[40];
    float* outp = (float*)d_out;

    static const int CIN[6]  = {1, 1024, 128, 128, 128, 256};
    static const int LIN[6]  = {1024, 128, 64, 32, 16, 8};
    static const int PADL[6] = {254, 31, 31, 31, 31, 31};
    static const int COUT[6] = {1024, 128, 128, 128, 256, 512};
    static const int LOUT[6] = {256, 128, 64, 32, 16, 8};
    static const int WSZ[6]  = {1024*1*512, 128*1024*64, 128*128*64,
                                128*128*64, 256*128*64, 512*256*64};
    static const int ASZ[6]  = {N_FRAMES*1024*128, N_FRAMES*128*64, N_FRAMES*128*32,
                                N_FRAMES*128*16,  N_FRAMES*256*8,  N_FRAMES*512*4};

    char* p = (char*)d_ws;
    auto carve = [&](size_t bytes) -> char* {
        char* r = p;
        p += (bytes + 255) & ~size_t(255);
        return r;
    };
    bf16_t* frames = (bf16_t*)carve(size_t(N_FRAMES) * WINDOW_ * 2);
    bf16_t* wbf[6]; for (int i = 0; i < 6; ++i) wbf[i] = (bf16_t*)carve(size_t(WSZ[i]) * 2);
    bf16_t* act[6]; for (int i = 0; i < 6; ++i) act[i] = (bf16_t*)carve(size_t(ASZ[i]) * 2);
    float*  probs  = (float*)carve(size_t(N_FRAMES) * PITCH_BINS_ * 4);
    float*  wts    = (float*)carve(size_t(PITCH_BINS_) * 4);

    frames_kernel<<<N_FRAMES, 256, 0, stream>>>(audio, frames);

    for (int i = 0; i < 6; ++i)
        cvt_bf16_kernel<<<(WSZ[i] + 255) / 256, 256, 0, stream>>>(cw[i], wbf[i], WSZ[i]);

    const bf16_t* cur = frames;
    for (int i = 0; i < 6; ++i) {
        int M = N_FRAMES * LOUT[i];
        dim3 grid((M + 63) / 64, COUT[i] / 128);
        switch (i) {
        case 0: conv_wmma_kernel<9, 8, 4><<<grid, 256, 0, stream>>>(
                    cur, wbf[0], cb[0], bn_g[0], bn_be[0], bn_m[0], bn_v[0], act[0],
                    N_FRAMES, CIN[0], LIN[0], COUT[0], PADL[0]); break;
        case 1: conv_wmma_kernel<6, 7, 1><<<grid, 256, 0, stream>>>(
                    cur, wbf[1], cb[1], bn_g[1], bn_be[1], bn_m[1], bn_v[1], act[1],
                    N_FRAMES, CIN[1], LIN[1], COUT[1], PADL[1]); break;
        case 2: conv_wmma_kernel<6, 6, 1><<<grid, 256, 0, stream>>>(
                    cur, wbf[2], cb[2], bn_g[2], bn_be[2], bn_m[2], bn_v[2], act[2],
                    N_FRAMES, CIN[2], LIN[2], COUT[2], PADL[2]); break;
        case 3: conv_wmma_kernel<6, 5, 1><<<grid, 256, 0, stream>>>(
                    cur, wbf[3], cb[3], bn_g[3], bn_be[3], bn_m[3], bn_v[3], act[3],
                    N_FRAMES, CIN[3], LIN[3], COUT[3], PADL[3]); break;
        case 4: conv_wmma_kernel<6, 4, 1><<<grid, 256, 0, stream>>>(
                    cur, wbf[4], cb[4], bn_g[4], bn_be[4], bn_m[4], bn_v[4], act[4],
                    N_FRAMES, CIN[4], LIN[4], COUT[4], PADL[4]); break;
        case 5: conv_wmma_kernel<6, 3, 1><<<grid, 256, 0, stream>>>(
                    cur, wbf[5], cb[5], bn_g[5], bn_be[5], bn_m[5], bn_v[5], act[5],
                    N_FRAMES, CIN[5], LIN[5], COUT[5], PADL[5]); break;
        }
        cur = act[i];
    }

    classifier_kernel<<<(N_FRAMES * PITCH_BINS_ + 255) / 256, 256, 0, stream>>>(
        act[5], cls_w, cls_b, probs);

    dither_weights_kernel<<<2, 256, 0, stream>>>(wts);

    postproc_kernel<<<N_FRAMES, 64, 0, stream>>>(probs, wts, f0min, f0max, outp);
}
